// AnyprecisionLinear_5643587027570
// MI455X (gfx1250) — matrix operand order repaired and checked by reading, not compiled, not run on hardware
//
#include <hip/hip_runtime.h>
#include <hip/hip_bf16.h>
#include <stdint.h>

// ---------------------------------------------------------------------------
// AnyprecisionLinear bulk path for gfx1250 (MI455X):
//   out[M,N] = x[M,K] * W[N,K]^T,  W from 4-bit codes + per-row 16-entry LUT.
// Compute-bound (AI ~ 1430 FLOP/B): materialize bf16 x and bf16 W once in
// scratch (bandwidth-bound pre-passes, HBM has huge slack at 23.3 TB/s), then
// run a pure-bf16 V_WMMA_F32_16X16X32_BF16 GEMM. Tile staging uses CDNA5
// async global->LDS copies (ASYNCcnt) when the toolchain exposes them,
// eliminating the VGPR round-trip in the hot loop.
// ---------------------------------------------------------------------------

typedef __attribute__((ext_vector_type(16))) __bf16 v16bf;
typedef __attribute__((ext_vector_type(8)))  float  v8f;
typedef __attribute__((ext_vector_type(4)))  int    v4i;

constexpr int M = 2048;          // B*S tokens
constexpr int N = 8192;          // OUT features
constexpr int K = 8192;          // IN features
constexpr int KW = K / 4;        // int32 words per qweight row

constexpr int BM = 128;
constexpr int BN = 128;
constexpr int BK = 32;           // K per WMMA step (16x16x32 bf16)
constexpr int NTHREADS = 256;    // 8 wave32 waves
constexpr int KSTEPS = K / BK;

static_assert(M % BM == 0 && N % BN == 0 && K % BK == 0, "tiling");

#if __has_builtin(__builtin_amdgcn_global_load_async_to_lds_b128) && \
    __has_builtin(__builtin_amdgcn_s_wait_asynccnt)
#define ASYNC_OK 1
#else
#define ASYNC_OK 0
#endif

__device__ __forceinline__ unsigned short f32_to_bf16(float f) {
    union { float f; uint32_t u; } v; v.f = f;
    uint32_t r = v.u + 0x7FFFu + ((v.u >> 16) & 1u);   // RNE
    return (unsigned short)(r >> 16);
}

// Pack two f32 -> two bf16 in one dword with a single V_PERM_B32.
__device__ __forceinline__ uint32_t pack2_bf16(float a, float b) {
    uint32_t au = __float_as_uint(a) + 0x8000u;
    uint32_t bu = __float_as_uint(b) + 0x8000u;
    return __builtin_amdgcn_perm(bu, au, 0x07060302u);   // bytes: a2,a3,b2,b3
}

#if ASYNC_OK
typedef __attribute__((address_space(1))) v4i* gv4i_p;   // global int4*
typedef __attribute__((address_space(3))) v4i* lv4i_p;   // LDS int4*
__device__ __forceinline__ void async_cp_b128(const unsigned short* g, unsigned short* l) {
    __builtin_amdgcn_global_load_async_to_lds_b128(
        (gv4i_p)(void*)g, (lv4i_p)(void*)l, 0, 0);
}
#endif

union Frag { v16bf v; uint4 q[2]; };

// ---------------------------------------------------------------------------
// Pre-pass 1: x f32 -> bf16   (64 MB read, 32 MB write)
// ---------------------------------------------------------------------------
__global__ __launch_bounds__(NTHREADS)
void convert_x_bf16(const float* __restrict__ x, unsigned short* __restrict__ xb) {
    size_t base = ((size_t)blockIdx.x * NTHREADS + threadIdx.x) * 8;   // 8 f32 / thread
    float4 a = *(const float4*)(x + base);
    float4 b = *(const float4*)(x + base + 4);
    uint4 r;
    r.x = pack2_bf16(a.x, a.y);
    r.y = pack2_bf16(a.z, a.w);
    r.z = pack2_bf16(b.x, b.y);
    r.w = pack2_bf16(b.z, b.w);
    *(uint4*)&xb[base] = r;
}

// ---------------------------------------------------------------------------
// Pre-pass 2: dequant W -> bf16 (N,K) row-major  (64 MB read, 128 MB write)
// ---------------------------------------------------------------------------
__global__ __launch_bounds__(NTHREADS)
void dequant_w_bf16(const int* __restrict__ qweight, const float* __restrict__ lut,
                    unsigned short* __restrict__ wb) {
    __shared__ unsigned short lrow[16];
    const int o = blockIdx.x;
    if (threadIdx.x < 16)
        lrow[threadIdx.x] = f32_to_bf16(lut[(size_t)o * 16 + threadIdx.x]);
    __syncthreads();
    const int* q = qweight + (size_t)o * KW;
    unsigned short* w = wb + (size_t)o * K;
    for (int i = threadIdx.x; i < KW; i += NTHREADS) {
        uint32_t wv = (uint32_t)q[i];
        uint32_t c0 = (wv >> 4)  & 0xF, c1 = (wv >> 12) & 0xF;
        uint32_t c2 = (wv >> 20) & 0xF, c3 = (wv >> 28) & 0xF;
        uint32_t lo = (uint32_t)lrow[c0] | ((uint32_t)lrow[c1] << 16);
        uint32_t hi = (uint32_t)lrow[c2] | ((uint32_t)lrow[c3] << 16);
        *(uint2*)&w[(size_t)i * 4] = make_uint2(lo, hi);
    }
}

// ---------------------------------------------------------------------------
// GEMM. PRE=true: x and W pre-materialized as bf16 (lean inner loop, async
// global->LDS staging when available). PRE=false: fused dequant fallback.
// ---------------------------------------------------------------------------
template <bool PRE>
__global__ __launch_bounds__(NTHREADS, 2)
void apq_gemm_bf16_wmma(const float* __restrict__ x,
                        const int*   __restrict__ qweight,
                        const float* __restrict__ lut,
                        const unsigned short* __restrict__ xb,
                        const unsigned short* __restrict__ wb,
                        float* __restrict__ out)
{
    __shared__ __align__(16) unsigned short Alds[2][BM * BK];        // 16 KB
    __shared__ __align__(16) unsigned short Blds[2][BN * BK];        // 16 KB
    __shared__ __align__(16) unsigned short LutLds[PRE ? 16 : BN * 16];

    const int tid  = threadIdx.x;
    const int lane = tid & 31;
    const int wave = tid >> 5;
    const int wm   = wave >> 2;          // 0..1  (M, 64 rows each)
    const int wn   = wave & 3;           // 0..3  (N, 32 cols each)

    const int m0 = blockIdx.y * BM;
    const int n0 = blockIdx.x * BN;

    // fallback-path per-thread mappings
    const int browq = tid >> 1;
    const int bwb   = (tid & 1) * 4;
    const int* qrow = qweight + (size_t)(n0 + browq) * KW;

    if constexpr (!PRE) {
        for (int i = tid; i < BN * 16; i += NTHREADS)
            LutLds[i] = f32_to_bf16(lut[(size_t)(n0 + (i >> 4)) * 16 + (i & 15)]);
        __syncthreads();
    }

    v8f zero = {0.f, 0.f, 0.f, 0.f, 0.f, 0.f, 0.f, 0.f};
    v8f acc[4][2];
#pragma unroll
    for (int mt = 0; mt < 4; ++mt)
#pragma unroll
        for (int nt = 0; nt < 2; ++nt) acc[mt][nt] = zero;

    // PRE-path per-thread mapping: 512 uint4 chunks per tile, 2 per thread
    // chunk c: row = c>>2, kc = (c&3)*8 bf16 elements

    // ---- prologue: stage K-step 0 into LDS buffer 0 ----
    if constexpr (PRE) {
#pragma unroll
        for (int i = 0; i < 2; ++i) {
            int c = tid + NTHREADS * i;
            int row = c >> 2, kc = (c & 3) * 8;
#if ASYNC_OK
            async_cp_b128(xb + (size_t)(m0 + row) * K + kc, &Alds[0][row * BK + kc]);
            async_cp_b128(wb + (size_t)(n0 + row) * K + kc, &Blds[0][row * BK + kc]);
#else
            uint4 a = *(const uint4*)(xb + (size_t)(m0 + row) * K + kc);
            uint4 b = *(const uint4*)(wb + (size_t)(n0 + row) * K + kc);
            *(uint4*)&Alds[0][row * BK + kc] = a;
            *(uint4*)&Blds[0][row * BK + kc] = b;
#endif
        }
    } else {
#pragma unroll
        for (int i = 0; i < 4; ++i) {
            int c = tid + NTHREADS * i;
            int row = c >> 3, kc = (c & 7) << 2;
            float4 f4 = *(const float4*)(x + (size_t)(m0 + row) * K + kc);
            *(uint2*)&Alds[0][row * BK + kc] =
                make_uint2(pack2_bf16(f4.x, f4.y), pack2_bf16(f4.z, f4.w));
        }
#pragma unroll
        for (int i = 0; i < 4; ++i) {
            uint32_t wv = (uint32_t)qrow[bwb + i];
            uint32_t c0 = (wv >> 4)  & 0xF, c1 = (wv >> 12) & 0xF;
            uint32_t c2 = (wv >> 20) & 0xF, c3 = (wv >> 28) & 0xF;
            const unsigned short* lr = &LutLds[browq * 16];
            *(uint2*)&Blds[0][browq * BK + (bwb + i) * 4] =
                make_uint2((uint32_t)lr[c0] | ((uint32_t)lr[c1] << 16),
                           (uint32_t)lr[c2] | ((uint32_t)lr[c3] << 16));
        }
    }

    // ---- main K loop, double buffered, one barrier per step ----
    for (int kt = 0; kt < KSTEPS; ++kt) {
#if ASYNC_OK
        if constexpr (PRE) __builtin_amdgcn_s_wait_asynccnt(0);
#endif
        __syncthreads();
        const int cur = kt & 1, nxt = cur ^ 1;
        const bool haveNext = (kt + 1) < KSTEPS;
        const int knext = (kt + 1) * BK;

        // issue next-stage staging
        uint4    aN[2], bN[2];      // PRE non-async path
        float4   af[4];             // fallback path
        uint32_t bw[4];
        if (haveNext) {
            if constexpr (PRE) {
#pragma unroll
                for (int i = 0; i < 2; ++i) {
                    int c = tid + NTHREADS * i;
                    int row = c >> 2, kc = (c & 3) * 8;
#if ASYNC_OK
                    // async copy engine fills the other buffer while we WMMA
                    async_cp_b128(xb + (size_t)(m0 + row) * K + knext + kc,
                                  &Alds[nxt][row * BK + kc]);
                    async_cp_b128(wb + (size_t)(n0 + row) * K + knext + kc,
                                  &Blds[nxt][row * BK + kc]);
#else
                    aN[i] = *(const uint4*)(xb + (size_t)(m0 + row) * K + knext + kc);
                    bN[i] = *(const uint4*)(wb + (size_t)(n0 + row) * K + knext + kc);
#endif
                }
            } else {
#pragma unroll
                for (int i = 0; i < 4; ++i) {
                    int c = tid + NTHREADS * i;
                    int row = c >> 3, kc = (c & 7) << 2;
                    af[i] = *(const float4*)(x + (size_t)(m0 + row) * K + knext + kc);
                }
#pragma unroll
                for (int i = 0; i < 4; ++i)
                    bw[i] = (uint32_t)qrow[(knext >> 2) + bwb + i];
            }
        }
        if (kt + 2 < KSTEPS) {   // speculative L2 prefetch two stages ahead
            if constexpr (PRE) {
                __builtin_prefetch(xb + (size_t)(m0 + (tid >> 2)) * K + (kt + 2) * BK, 0, 1);
                __builtin_prefetch(wb + (size_t)(n0 + (tid >> 2)) * K + (kt + 2) * BK, 0, 1);
            } else {
                __builtin_prefetch(x + (size_t)(m0 + (tid >> 3)) * K + (kt + 2) * BK, 0, 1);
                __builtin_prefetch(qweight + (size_t)(n0 + browq) * KW + (((kt + 2) * BK) >> 2), 0, 1);
            }
        }

        // B fragments: lane -> col n = lane&15, K = (lane>=16 ? 16..31 : 0..15)
        v16bf bmat[2];
#pragma unroll
        for (int nt = 0; nt < 2; ++nt) {
            int n  = wn * 32 + nt * 16 + (lane & 15);
            int kb = (lane >> 4) * 16;
            Frag f;
            f.q[0] = *(const uint4*)&Blds[cur][n * BK + kb];
            f.q[1] = *(const uint4*)&Blds[cur][n * BK + kb + 8];
            bmat[nt] = f.v;
        }

        // A fragments + WMMA: lane -> row m = lane&15,
        // K = (lane>=16 ? 16..31 : 0..15)
#pragma unroll
        for (int mt = 0; mt < 4; ++mt) {
            int row = wm * 64 + mt * 16 + (lane & 15);
            int kb  = (lane >> 4) * 16;
            Frag f;
            f.q[0] = *(const uint4*)&Alds[cur][row * BK + kb];
            f.q[1] = *(const uint4*)&Alds[cur][row * BK + kb + 8];
#pragma unroll
            for (int nt = 0; nt < 2; ++nt) {
                acc[mt][nt] = __builtin_amdgcn_wmma_f32_16x16x32_bf16(
                    false, f.v, false, bmat[nt], (short)0, acc[mt][nt],
                    false, false);
            }
        }

        // commit next stage (only paths that staged through VGPRs)
        if (haveNext) {
            if constexpr (PRE) {
#if !ASYNC_OK
#pragma unroll
                for (int i = 0; i < 2; ++i) {
                    int c = tid + NTHREADS * i;
                    int row = c >> 2, kc = (c & 3) * 8;
                    *(uint4*)&Alds[nxt][row * BK + kc] = aN[i];
                    *(uint4*)&Blds[nxt][row * BK + kc] = bN[i];
                }
#endif
            } else {
#pragma unroll
                for (int i = 0; i < 4; ++i) {
                    int c = tid + NTHREADS * i;
                    int row = c >> 3, kc = (c & 7) << 2;
                    *(uint2*)&Alds[nxt][row * BK + kc] =
                        make_uint2(pack2_bf16(af[i].x, af[i].y), pack2_bf16(af[i].z, af[i].w));
                }
#pragma unroll
                for (int i = 0; i < 4; ++i) {
                    uint32_t wv = bw[i];
                    uint32_t c0 = (wv >> 4)  & 0xF, c1 = (wv >> 12) & 0xF;
                    uint32_t c2 = (wv >> 20) & 0xF, c3 = (wv >> 28) & 0xF;
                    const unsigned short* lr = &LutLds[browq * 16];
                    *(uint2*)&Blds[nxt][browq * BK + (bwb + i) * 4] =
                        make_uint2((uint32_t)lr[c0] | ((uint32_t)lr[c1] << 16),
                                   (uint32_t)lr[c2] | ((uint32_t)lr[c3] << 16));
                }
            }
        }
    }

    // ---- epilogue: C layout VGPR r -> (M = r + 8*(lane>=16), N = lane&15) ----
#pragma unroll
    for (int mt = 0; mt < 4; ++mt) {
#pragma unroll
        for (int nt = 0; nt < 2; ++nt) {
            int n     = n0 + wn * 32 + nt * 16 + (lane & 15);
            int mbase = m0 + wm * 64 + mt * 16 + (lane >> 4) * 8;
#pragma unroll
            for (int r = 0; r < 8; ++r)
                out[(size_t)(mbase + r) * N + n] = acc[mt][nt][r];
        }
    }
}

extern "C" void kernel_launch(void* const* d_in, const int* in_sizes, int n_in,
                              void* d_out, int out_size, void* d_ws, size_t ws_size,
                              hipStream_t stream) {
    const float* x   = (const float*)d_in[0];   // (1, 2048, 8192) f32
    const int*   qw  = (const int*)d_in[1];     // (8192, 2048) int32
    const float* lut = (const float*)d_in[2];   // (8192, 16) f32
    float*       out = (float*)d_out;           // (1, 2048, 8192) f32

    const size_t xbytes = (size_t)M * K * sizeof(unsigned short);   // 32 MB
    const size_t wbytes = (size_t)N * K * sizeof(unsigned short);   // 128 MB

    dim3 grid(N / BN, M / BM);   // 64 x 16 blocks
    dim3 block(NTHREADS);

    if (ws_size >= xbytes + wbytes) {
        unsigned short* xb = (unsigned short*)d_ws;
        unsigned short* wb = (unsigned short*)((char*)d_ws + xbytes);
        convert_x_bf16<<<(M * K) / (8 * NTHREADS), block, 0, stream>>>(x, xb);
        dequant_w_bf16<<<N, block, 0, stream>>>(qw, lut, wb);
        apq_gemm_bf16_wmma<true><<<grid, block, 0, stream>>>(x, qw, lut, xb, wb, out);
    } else {
        apq_gemm_bf16_wmma<false><<<grid, block, 0, stream>>>(x, qw, lut, nullptr, nullptr, out);
    }
}
